// SwinTransformer_18794776887758
// MI455X (gfx1250) — compile-verified, matrix operand order checked by reading, not run on hardware
//
#include <hip/hip_runtime.h>
#include <hip/hip_bf16.h>
#include <cstdint>

// MI455X / gfx1250, wave32. All GEMM-shaped work via v_wmma_f32_16x16x32_f16.
// Workspace layout (bytes):
//   [0, 100663296)              : xw   f16 [262144 x 192]   (reused as attn_out)
//   [100663296, 402653184)      : qkv  f16 [262144 x 576]
// => needs 384 MiB of d_ws.

typedef __attribute__((ext_vector_type(16))) _Float16 v16h;
typedef __attribute__((ext_vector_type(8)))  float    v8f;

#define SSZ 4
#define NHD 6
#define DIMC 192
#define HD 32
#define QKV_N 576
#define HID 768
#define ROWS_TOTAL 262144  // 4 * 256 * 256

union AF { v16h v; uint32_t u[8]; _Float16 h[16]; };

// K-pair start for fragment register r (0..7), lane-group g (0/1).
// A/B 16-bit layout: VGPR0..3 -> K {0..7}+8g, VGPR4..7 -> K {16..23}+8g.
__device__ __forceinline__ int kpair_idx(int r, int g) {
    return ((r >> 2) << 4) | (g << 3) | ((r & 3) << 1);
}

// Gather a 16x32 f16 fragment (A: rows=M, or B stored [N][K]: rows=N) from a
// row-major tile. K-pairs are contiguous -> one dword load per register.
__device__ __forceinline__ v16h wfrag_rm(const _Float16* tile, int stride, int kbase) {
    int lane = threadIdx.x & 31;
    int row = lane & 15, g = lane >> 4;
    AF f;
    const _Float16* rp = tile + row * stride + kbase;
#pragma unroll
    for (int r = 0; r < 8; ++r)
        f.u[r] = *(const uint32_t*)(rp + kpair_idx(r, g));
    return f.v;
}

__device__ __forceinline__ v8f wmma32(v16h a, v16h b, v8f c) {
    return __builtin_amdgcn_wmma_f32_16x16x32_f16(false, a, false, b, (short)0, c, false, false);
}

// ---------------------------------------------------------------------------
// Kernel 1: LayerNorm1 + cyclic shift (-4,-4) + window partition -> f16.
// One wave per token (lane handles 6 of 192 channels).
// ---------------------------------------------------------------------------
__global__ __launch_bounds__(256) void k_ln_shift_window(
    const float* __restrict__ x, const float* __restrict__ w,
    const float* __restrict__ b, _Float16* __restrict__ xw)
{
    int lane = threadIdx.x & 31;
    int tok = blockIdx.x * 8 + (threadIdx.x >> 5);   // windowed row index
    int bidx = tok >> 16;
    int rem  = tok & 65535;
    int wIdx = rem >> 6, off = rem & 63;
    int wh = wIdx >> 5, ww = wIdx & 31;
    int i = off >> 3, j = off & 7;
    int h  = (wh * 8 + i + SSZ) & 255;               // shifted-gather source
    int w_ = (ww * 8 + j + SSZ) & 255;
    const float* xp = x + ((size_t)bidx * 65536 + h * 256 + w_) * DIMC + lane * 6;
    float v0[6]; float s = 0.f;
#pragma unroll
    for (int e = 0; e < 6; ++e) { v0[e] = xp[e]; s += v0[e]; }
#pragma unroll
    for (int m = 1; m < 32; m <<= 1) s += __shfl_xor(s, m, 32);
    float mu = s * (1.0f / 192.0f);
    float var = 0.f;
#pragma unroll
    for (int e = 0; e < 6; ++e) { float d = v0[e] - mu; var += d * d; }
#pragma unroll
    for (int m = 1; m < 32; m <<= 1) var += __shfl_xor(var, m, 32);
    float rs = rsqrtf(var * (1.0f / 192.0f) + 1e-5f);
    _Float16* op = xw + (size_t)tok * DIMC + lane * 6;
#pragma unroll
    for (int e = 0; e < 6; ++e) {
        int c = lane * 6 + e;
        op[e] = (_Float16)((v0[e] - mu) * rs * w[c] + b[c]);
    }
}

// ---------------------------------------------------------------------------
// Kernel 2: tiled WMMA GEMM  out[M,N] = A[M,K](f16) @ W[N,K]^T(f32) + bias
// 128 threads = 4 waves; 64x64 tile; wave w owns rows w*16..w*16+15.
// mode 0: store f16 (QKV).  mode 1: window-reverse scatter + residual -> f32.
// ---------------------------------------------------------------------------
__global__ __launch_bounds__(128) void k_gemm(
    const _Float16* __restrict__ A, const float* __restrict__ W,
    const float* __restrict__ bias, int K, int Ntot,
    _Float16* __restrict__ outH, float* __restrict__ outF,
    const float* __restrict__ res, int mode)
{
    __shared__ __align__(16) _Float16 ldsA[64][40];
    __shared__ __align__(16) _Float16 ldsB[64][40];
    int tid = threadIdx.x;
    int wid = tid >> 5, lane = tid & 31;
    int g = lane >> 4, l16 = lane & 15;
    size_t m0 = (size_t)blockIdx.x * 64;
    int n0 = blockIdx.y * 64;
    v8f acc[4] = {};

    for (int k0 = 0; k0 < K; k0 += 32) {
#pragma unroll
        for (int it = 0; it < 8; ++it) {              // A: 1024 dwords
            int i = tid * 8 + it;
            int m = i >> 4, kk = (i & 15) << 1;
            *(uint32_t*)&ldsA[m][kk] =
                *(const uint32_t*)(A + (m0 + m) * (size_t)K + k0 + kk);
        }
#pragma unroll
        for (int it = 0; it < 16; ++it) {             // W: 2048 f32->f16
            int i = tid * 16 + it;
            int n = i >> 5, kk = i & 31;
            ldsB[n][kk] = (_Float16)W[(size_t)(n0 + n) * K + k0 + kk];
        }
        __syncthreads();
        v16h a = wfrag_rm(&ldsA[wid * 16][0], 40, 0);
#pragma unroll
        for (int t = 0; t < 4; ++t) {
            v16h bf = wfrag_rm(&ldsB[t * 16][0], 40, 0);
            acc[t] = wmma32(a, bf, acc[t]);
        }
        __syncthreads();
    }

    if (mode == 0) {
#pragma unroll
        for (int t = 0; t < 4; ++t)
#pragma unroll
            for (int r = 0; r < 8; ++r) {
                size_t m = m0 + wid * 16 + r + 8 * g;
                int n = n0 + t * 16 + l16;
                outH[m * Ntot + n] = (_Float16)(acc[t][r] + bias[n]);
            }
    } else {   // window-reverse + unshift (+4,+4) scatter, residual add
#pragma unroll
        for (int t = 0; t < 4; ++t)
#pragma unroll
            for (int r = 0; r < 8; ++r) {
                size_t m = m0 + wid * 16 + r + 8 * g;
                int n = n0 + t * 16 + l16;
                int bi = (int)(m >> 16);
                int rem = (int)m & 65535;
                int wIdx = rem >> 6, off = rem & 63;
                int wh = wIdx >> 5, ww = wIdx & 31;
                int i2 = off >> 3, j2 = off & 7;
                int h  = (wh * 8 + i2 + SSZ) & 255;
                int w2 = (ww * 8 + j2 + SSZ) & 255;
                size_t dst = ((size_t)bi * 65536 + h * 256 + w2) * DIMC + n;
                outF[dst] = acc[t][r] + bias[n] + res[dst];
            }
    }
}

// ---------------------------------------------------------------------------
// Kernel 3: windowed attention. One wave per (window, head). hd=32 => QK^T is
// exactly one 16x16x32 WMMA per tile. Bias+shift-mask computed analytically.
// ---------------------------------------------------------------------------
__global__ __launch_bounds__(128) void k_attn(
    const _Float16* __restrict__ qkv, const float* __restrict__ rel_bias,
    _Float16* __restrict__ attn_out)
{
    __shared__ __align__(16) _Float16 ldsP[4][16][72];
    int wid = threadIdx.x >> 5, lane = threadIdx.x & 31;
    int g = lane >> 4, l16 = lane & 15;
    int task = blockIdx.x * 4 + wid;
    int win = task / NHD, head = task % NHD;
    int wIdx = win & 1023;
    int wh = wIdx >> 5, ww = wIdx & 31;
    const _Float16* qp = qkv + (size_t)win * 64 * QKV_N + head * HD;
    const _Float16* kp = qp + DIMC;
    const _Float16* vp = qp + 2 * DIMC;

    // K^T fragments (B operand): tile ni -> key token = ni*16 + l16, K dim = d
    v16h kB[4];
#pragma unroll
    for (int ni = 0; ni < 4; ++ni) {
        AF f;
        const _Float16* rp = kp + (size_t)(ni * 16 + l16) * QKV_N;
#pragma unroll
        for (int r = 0; r < 8; ++r)
            f.u[r] = *(const uint32_t*)(rp + kpair_idx(r, g));
        kB[ni] = f.v;
    }
    // V fragments (B operand): K dim = key token (stride 576), N = d
    v16h vB[4];
#pragma unroll
    for (int kt = 0; kt < 2; ++kt)
#pragma unroll
        for (int nt = 0; nt < 2; ++nt) {
            AF f;
            int d = nt * 16 + l16;
#pragma unroll
            for (int r = 0; r < 8; ++r) {
                int kk = kt * 32 + kpair_idx(r, g);
                f.h[2 * r]     = vp[(size_t)kk * QKV_N + d];
                f.h[2 * r + 1] = vp[(size_t)(kk + 1) * QKV_N + d];
            }
            vB[kt * 2 + nt] = f.v;
        }

    const float scale = 0.17677669529663687f;   // 1/sqrt(32)
    for (int mi = 0; mi < 4; ++mi) {
        AF qa;
        const _Float16* rp = qp + (size_t)(mi * 16 + l16) * QKV_N;
#pragma unroll
        for (int r = 0; r < 8; ++r)
            qa.u[r] = *(const uint32_t*)(rp + kpair_idx(r, g));

        v8f p[4] = {};
#pragma unroll
        for (int ni = 0; ni < 4; ++ni)
            p[ni] = wmma32(qa.v, kB[ni], p[ni]);

        // scale + relative-position bias + shift-window mask
#pragma unroll
        for (int ni = 0; ni < 4; ++ni)
#pragma unroll
            for (int r = 0; r < 8; ++r) {
                int n = mi * 16 + r + 8 * g;         // query token in window
                int m = ni * 16 + l16;               // key token in window
                int i1 = n >> 3, j1 = n & 7, i2 = m >> 3, j2 = m & 7;
                int idx = (i1 - i2 + 7) * 15 + (j1 - j2 + 7);
                float bv = rel_bias[idx * NHD + head];
                int gh1 = wh * 8 + i1, gw1 = ww * 8 + j1;
                int gh2 = wh * 8 + i2, gw2 = ww * 8 + j2;
                int r1 = (gh1 < 248 ? 0 : (gh1 < 252 ? 1 : 2)) * 3 +
                         (gw1 < 248 ? 0 : (gw1 < 252 ? 1 : 2));
                int r2 = (gh2 < 248 ? 0 : (gh2 < 252 ? 1 : 2)) * 3 +
                         (gw2 < 248 ? 0 : (gw2 < 252 ? 1 : 2));
                float mk = (r1 != r2) ? -100.0f : 0.0f;
                p[ni][r] = p[ni][r] * scale + bv + mk;
            }

        // row softmax: each VGPR r holds one row across a 16-lane half-wave
#pragma unroll
        for (int r = 0; r < 8; ++r) {
            float mx = fmaxf(fmaxf(p[0][r], p[1][r]), fmaxf(p[2][r], p[3][r]));
#pragma unroll
            for (int o = 1; o < 16; o <<= 1) mx = fmaxf(mx, __shfl_xor(mx, o, 16));
            float sm = 0.f;
#pragma unroll
            for (int ni = 0; ni < 4; ++ni) { p[ni][r] = __expf(p[ni][r] - mx); sm += p[ni][r]; }
#pragma unroll
            for (int o = 1; o < 16; o <<= 1) sm += __shfl_xor(sm, o, 16);
            float inv = 1.0f / sm;
#pragma unroll
            for (int ni = 0; ni < 4; ++ni) p[ni][r] *= inv;
        }

        // LDS round-trip: C-layout -> A-operand layout
#pragma unroll
        for (int ni = 0; ni < 4; ++ni)
#pragma unroll
            for (int r = 0; r < 8; ++r)
                ldsP[wid][r + 8 * g][ni * 16 + l16] = (_Float16)p[ni][r];
        __syncthreads();

        v8f o0 = {}, o1 = {};
#pragma unroll
        for (int kt = 0; kt < 2; ++kt) {
            v16h aa = wfrag_rm(&ldsP[wid][0][0], 72, kt * 32);
            o0 = wmma32(aa, vB[kt * 2 + 0], o0);
            o1 = wmma32(aa, vB[kt * 2 + 1], o1);
        }
#pragma unroll
        for (int r = 0; r < 8; ++r) {
            size_t row = (size_t)win * 64 + mi * 16 + r + 8 * g;
            int c0 = head * HD + l16;
            attn_out[row * DIMC + c0]      = (_Float16)o0[r];
            attn_out[row * DIMC + c0 + 16] = (_Float16)o1[r];
        }
        __syncthreads();
    }
}

// ---------------------------------------------------------------------------
// Kernel 4: fused LN2 + fc1 + GELU + fc2 + residual, in-place on d_out.
// 256 threads = 8 waves (4 M-groups x 2 N-groups); 64 rows per block.
// h kept in 128-col chunks in LDS; fc2 accumulated in registers.
// ---------------------------------------------------------------------------
__global__ __launch_bounds__(256) void k_mlp(
    float* __restrict__ xio,
    const float* __restrict__ n2w, const float* __restrict__ n2b,
    const float* __restrict__ w1, const float* __restrict__ b1,
    const float* __restrict__ w2, const float* __restrict__ b2)
{
    __shared__ __align__(16) _Float16 ldsX[64][200];  // LN2 output (f16)
    __shared__ __align__(16) _Float16 ldsH[64][136];  // gelu(h) chunk
    __shared__ __align__(16) _Float16 ldsW[192][40];  // staged weight tile
    int tid = threadIdx.x;
    int wid = tid >> 5, lane = tid & 31;
    int g = lane >> 4, l16 = lane & 15;
    size_t row0 = (size_t)blockIdx.x * 64;

    // LayerNorm2 -> ldsX (each wave: 8 rows; lane: 6 channels)
    for (int rr = 0; rr < 8; ++rr) {
        int row = wid * 8 + rr;
        const float* xp = xio + (row0 + row) * DIMC + lane * 6;
        float v0[6]; float s = 0.f;
#pragma unroll
        for (int e = 0; e < 6; ++e) { v0[e] = xp[e]; s += v0[e]; }
#pragma unroll
        for (int m = 1; m < 32; m <<= 1) s += __shfl_xor(s, m, 32);
        float mu = s * (1.f / 192.f);
        float var = 0.f;
#pragma unroll
        for (int e = 0; e < 6; ++e) { float d = v0[e] - mu; var += d * d; }
#pragma unroll
        for (int m = 1; m < 32; m <<= 1) var += __shfl_xor(var, m, 32);
        float rs = rsqrtf(var * (1.f / 192.f) + 1e-5f);
#pragma unroll
        for (int e = 0; e < 6; ++e) {
            int c = lane * 6 + e;
            ldsX[row][c] = (_Float16)((v0[e] - mu) * rs * n2w[c] + n2b[c]);
        }
    }
    __syncthreads();

    int wm = wid & 3, wn = wid >> 2;
    v8f oacc[6] = {};
    for (int kc = 0; kc < 6; ++kc) {                  // 6 chunks of 128 h-cols
        v8f hacc[4] = {};
        for (int ks = 0; ks < 6; ++ks) {              // K=192 in steps of 32
#pragma unroll
            for (int it = 0; it < 16; ++it) {         // stage fc1_w [128n][32k]
                int i = tid * 16 + it;
                int n = i >> 5, kk = i & 31;
                ldsW[n][kk] = (_Float16)w1[(size_t)(kc * 128 + n) * DIMC + ks * 32 + kk];
            }
            __syncthreads();
            v16h a = wfrag_rm(&ldsX[wm * 16][0], 200, ks * 32);
#pragma unroll
            for (int nt = 0; nt < 4; ++nt) {
                v16h bf = wfrag_rm(&ldsW[wn * 64 + nt * 16][0], 40, 0);
                hacc[nt] = wmma32(a, bf, hacc[nt]);
            }
            __syncthreads();
        }
        // bias + exact GELU -> ldsH
#pragma unroll
        for (int nt = 0; nt < 4; ++nt)
#pragma unroll
            for (int r = 0; r < 8; ++r) {
                int cl = wn * 64 + nt * 16 + l16;
                float hv = hacc[nt][r] + b1[kc * 128 + cl];
                float ge = 0.5f * hv * (1.0f + erff(hv * 0.70710678118654752f));
                ldsH[wm * 16 + r + 8 * g][cl] = (_Float16)ge;
            }
        __syncthreads();
        // fc2 partial product over this chunk
        for (int k2 = 0; k2 < 4; ++k2) {
#pragma unroll
            for (int it = 0; it < 24; ++it) {         // stage fc2_w [192n][32k]
                int i = tid * 24 + it;
                int n = i >> 5, kk = i & 31;
                ldsW[n][kk] = (_Float16)w2[(size_t)n * HID + kc * 128 + k2 * 32 + kk];
            }
            __syncthreads();
            v16h a = wfrag_rm(&ldsH[wm * 16][0], 136, k2 * 32);
#pragma unroll
            for (int nt = 0; nt < 6; ++nt) {
                v16h bf = wfrag_rm(&ldsW[wn * 96 + nt * 16][0], 40, 0);
                oacc[nt] = wmma32(a, bf, oacc[nt]);
            }
            __syncthreads();
        }
    }
    // out = x1 + fc2 + b2 (in-place; block owns its 64 rows exclusively)
#pragma unroll
    for (int nt = 0; nt < 6; ++nt)
#pragma unroll
        for (int r = 0; r < 8; ++r) {
            size_t row = row0 + wm * 16 + r + 8 * g;
            int n = wn * 96 + nt * 16 + l16;
            float* op = xio + row * DIMC + n;
            *op = oacc[nt][r] + b2[n] + *op;
        }
}

// ---------------------------------------------------------------------------
extern "C" void kernel_launch(void* const* d_in, const int* in_sizes, int n_in,
                              void* d_out, int out_size, void* d_ws, size_t ws_size,
                              hipStream_t stream)
{
    (void)in_sizes; (void)n_in; (void)out_size; (void)ws_size;
    const float* x      = (const float*)d_in[0];
    const float* n1w    = (const float*)d_in[1];
    const float* n1b    = (const float*)d_in[2];
    const float* qkv_w  = (const float*)d_in[3];
    const float* qkv_b  = (const float*)d_in[4];
    const float* relb   = (const float*)d_in[5];
    const float* proj_w = (const float*)d_in[6];
    const float* proj_b = (const float*)d_in[7];
    const float* n2w    = (const float*)d_in[8];
    const float* n2b    = (const float*)d_in[9];
    const float* w1     = (const float*)d_in[10];
    const float* b1     = (const float*)d_in[11];
    const float* w2     = (const float*)d_in[12];
    const float* b2     = (const float*)d_in[13];
    float* out = (float*)d_out;

    _Float16* xw       = (_Float16*)d_ws;
    _Float16* qkvb     = xw + (size_t)ROWS_TOTAL * DIMC;
    _Float16* attn_out = xw;   // xw dead after QKV GEMM; reuse region

    // 1) LN1 + shift + window partition -> f16 activations
    k_ln_shift_window<<<ROWS_TOTAL / 8, 256, 0, stream>>>(x, n1w, n1b, xw);
    // 2) QKV GEMM: [262144,192] @ [192,576]^T -> f16
    k_gemm<<<dim3(ROWS_TOTAL / 64, QKV_N / 64), 128, 0, stream>>>(
        xw, qkv_w, qkv_b, DIMC, QKV_N, qkvb, nullptr, nullptr, 0);
    // 3) attention: 4096 windows x 6 heads, one wave each (4 waves/block)
    k_attn<<<(ROWS_TOTAL / 64) * NHD / 4, 128, 0, stream>>>(qkvb, relb, attn_out);
    // 4) proj GEMM + window-reverse/unshift scatter + residual -> d_out (= x1)
    k_gemm<<<dim3(ROWS_TOTAL / 64, DIMC / 64), 128, 0, stream>>>(
        attn_out, proj_w, proj_b, DIMC, DIMC, nullptr, out, x, 1);
    // 5) fused LN2 + MLP + residual, in-place on d_out
    k_mlp<<<ROWS_TOTAL / 64, 256, 0, stream>>>(out, n2w, n2b, w1, b1, w2, b2);
}
